// GeoFNO_12936441495792
// MI455X (gfx1250) — matrix operand order repaired and checked by reading, not compiled, not run on hardware
//
#include <hip/hip_runtime.h>
#include <hip/hip_bf16.h>
#include <math.h>

// ---- problem constants (from reference) ----
#define NN 16384   // nodes
#define KM 512     // fourier modes
#define CC 128     // channels
#define LL 4

typedef __bf16 bf16;
typedef __attribute__((ext_vector_type(16))) __bf16 v16bf;
typedef __attribute__((ext_vector_type(8)))  __bf16 v8bf;
typedef __attribute__((ext_vector_type(8)))  float  v8f;

// ---------------- device helpers ----------------

__device__ __forceinline__ float gelu_f(float x) {
  // jax.nn.gelu default (approximate=True, tanh form)
  float x3 = x * x * x;
  return 0.5f * x * (1.0f + tanhf(0.7978845608028654f * (x + 0.044715f * x3)));
}

// A fragment: 16x32 bf16 tile of row-major A (lda halves per row) at (row0,k0).
// lanes 0-15: row M=lane, K{0..15} in halves0-15;
// lanes 16-31: K{16..31}.  -> two aligned 16B loads per lane.
__device__ __forceinline__ v16bf load_a_frag(const bf16* A, int lda, int row0, int k0, int lane) {
  int r  = row0 + (lane & 15);
  int kb = k0 + ((lane & 16) ? 16 : 0);
  const bf16* p = A + (size_t)r * lda + kb;
  v8bf lo = *(const v8bf*)(p);
  v8bf hi = *(const v8bf*)(p + 8);
  return __builtin_shufflevector(lo, hi, 0, 1, 2, 3, 4, 5, 6, 7,
                                 8, 9, 10, 11, 12, 13, 14, 15);
}

// B fragment from COLUMN-MAJOR storage Bt (column ng contiguous over kg, ldk = Kdim stride):
// lane holds column n0+(lane&15); lanes 0-15 K{k0..k0+15}, lanes 16-31 K{k0+16..k0+31}.
// -> one 32B contiguous, 32B-aligned load per lane.
__device__ __forceinline__ v16bf load_bt_frag(const bf16* Bt, int ldk, int k0, int n0, int lane) {
  int n  = n0 + (lane & 15);
  int kb = k0 + ((lane & 16) ? 16 : 0);
  return *(const v16bf*)(Bt + (size_t)n * ldk + kb);
}

__device__ __forceinline__ v8f wmma_bf16(v16bf a, v16bf b, v8f c) {
  return __builtin_amdgcn_wmma_f32_16x16x32_bf16(false, a, false, b, (short)0, c, false, false);
}

// ---------------- kernels ----------------

// temp = nodes . modes ; bc=cos*mask, bs=sin*mask
// bc_nm/bs_nm [NN,KM] plain (column-major B of backward GEMM)  -> direct coalesced stores
// wbcT/wbsnT [KM,NN] weighted (column-major B of forward GEMM) -> LDS-transposed stores
__global__ void basis_kernel(const float* __restrict__ nodes, const float* __restrict__ modes,
                             const float* __restrict__ mask, const float* __restrict__ wts,
                             bf16* bc_nm, bf16* bs_nm, bf16* wbcT, bf16* wbsnT) {
  __shared__ float sc[16][17];
  __shared__ float ss[16][17];
  int k = blockIdx.x * 16 + threadIdx.x;
  int n = blockIdx.y * 16 + threadIdx.y;
  float t = nodes[n * 2 + 0] * modes[k * 2 + 0] + nodes[n * 2 + 1] * modes[k * 2 + 1];
  float m = mask[n], w = wts[n];
  float c = cosf(t) * m;
  float s = sinf(t) * m;
  bc_nm[(size_t)n * KM + k] = (bf16)c;
  bs_nm[(size_t)n * KM + k] = (bf16)s;
  sc[threadIdx.y][threadIdx.x] = c * w;
  ss[threadIdx.y][threadIdx.x] = -s * w;
  __syncthreads();
  int kt = blockIdx.x * 16 + threadIdx.y;
  int nt = blockIdx.y * 16 + threadIdx.x;
  wbcT [(size_t)kt * NN + nt] = (bf16)sc[threadIdx.x][threadIdx.y];
  wbsnT[(size_t)kt * NN + nt] = (bf16)ss[threadIdx.x][threadIdx.y];
}

// lift: h[c,n] = fc0_b[c] + sum_i x[n,i]*fc0_w[c,i]
__global__ void lift_kernel(const float* __restrict__ x, const float* __restrict__ fc0w,
                            const float* __restrict__ fc0b, float* h32, bf16* h_cn, bf16* h_nc) {
  int n = blockIdx.x * blockDim.x + threadIdx.x;
  float x0 = x[n * 3 + 0], x1 = x[n * 3 + 1], x2 = x[n * 3 + 2];
#pragma unroll 4
  for (int c = 0; c < CC; ++c) {
    float v = fc0b[c] + x0 * fc0w[c * 3 + 0] + x1 * fc0w[c * 3 + 1] + x2 * fc0w[c * 3 + 2];
    h32[(size_t)c * NN + n]  = v;
    h_cn[(size_t)c * NN + n] = (bf16)v;
    h_nc[(size_t)n * CC + c] = (bf16)v;
  }
}

// x0[i] = sum_n h[i,n] * wts[n] * mask[n]
__global__ void x0_kernel(const float* __restrict__ h32, const float* __restrict__ wts,
                          const float* __restrict__ mask, float* x0v) {
  __shared__ float red[256];
  int i = blockIdx.x;
  float s = 0.f;
  for (int n = threadIdx.x; n < NN; n += 256)
    s += h32[(size_t)i * NN + n] * wts[n] * mask[n];
  red[threadIdx.x] = s;
  __syncthreads();
  for (int off = 128; off; off >>= 1) {
    if (threadIdx.x < off) red[threadIdx.x] += red[threadIdx.x + off];
    __syncthreads();
  }
  if (threadIdx.x == 0) x0v[i] = red[0];
}

// f0[o] = sum_i x0[i] * sp_w0[i,o,0]
__global__ void f0_kernel(const float* __restrict__ x0v, const float* __restrict__ w0, float* f0) {
  int o = threadIdx.x;
  float s = 0.f;
  for (int i = 0; i < CC; ++i) s += x0v[i] * w0[(size_t)i * CC + o];
  f0[o] = s;
}

__global__ void cvt_bf16_kernel(const float* __restrict__ src, bf16* dst, int n) {
  int t = blockIdx.x * blockDim.x + threadIdx.x;
  if (t < n) dst[t] = (bf16)src[t];
}

// forward transform: xc = h @ wbc, xs = h @ wbsn  ([128,NN]@[NN,KM] -> [128,KM])
// wave computes 16x32 for both outputs: 8 mtiles x 16 npairs = 128 waves.
// software-pipelined: next K-step fragments prefetched before current WMMAs.
__global__ void gemm_fwd_kernel(const bf16* __restrict__ h_cn, const bf16* __restrict__ wbcT,
                                const bf16* __restrict__ wbsnT, float* xc, float* xs) {
  int wave = (blockIdx.x * blockDim.x + threadIdx.x) >> 5;
  int lane = threadIdx.x & 31;
  int m0 = (wave >> 4) * 16;
  int n0 = (wave & 15) * 32;
  v8f ac0 = {}, ac1 = {}, as0 = {}, as1 = {};

  v16bf a   = load_a_frag(h_cn, NN, m0, 0, lane);
  v16bf bc0 = load_bt_frag(wbcT,  NN, 0, n0,      lane);
  v16bf bc1 = load_bt_frag(wbcT,  NN, 0, n0 + 16, lane);
  v16bf bs0 = load_bt_frag(wbsnT, NN, 0, n0,      lane);
  v16bf bs1 = load_bt_frag(wbsnT, NN, 0, n0 + 16, lane);
  for (int k0 = 32; k0 < NN; k0 += 32) {
    v16bf aN  = load_a_frag(h_cn, NN, m0, k0, lane);
    v16bf nc0 = load_bt_frag(wbcT,  NN, k0, n0,      lane);
    v16bf nc1 = load_bt_frag(wbcT,  NN, k0, n0 + 16, lane);
    v16bf ns0 = load_bt_frag(wbsnT, NN, k0, n0,      lane);
    v16bf ns1 = load_bt_frag(wbsnT, NN, k0, n0 + 16, lane);
    ac0 = wmma_bf16(a, bc0, ac0);
    ac1 = wmma_bf16(a, bc1, ac1);
    as0 = wmma_bf16(a, bs0, as0);
    as1 = wmma_bf16(a, bs1, as1);
    a = aN; bc0 = nc0; bc1 = nc1; bs0 = ns0; bs1 = ns1;
  }
  ac0 = wmma_bf16(a, bc0, ac0);
  ac1 = wmma_bf16(a, bc1, ac1);
  as0 = wmma_bf16(a, bs0, as0);
  as1 = wmma_bf16(a, bs1, as1);

  int col = n0 + (lane & 15);
  int rb  = m0 + ((lane & 16) ? 8 : 0);
#pragma unroll
  for (int j = 0; j < 8; ++j) {
    xc[(size_t)(rb + j) * KM + col]      = ac0[j];
    xc[(size_t)(rb + j) * KM + col + 16] = ac1[j];
    xs[(size_t)(rb + j) * KM + col]      = as0[j];
    xs[(size_t)(rb + j) * KM + col + 16] = as1[j];
  }
}

// per-mode channel mixing: fc2 = 2*(xc@wc - xs@ws), fs2 = -2*(xs@wc + xc@ws)
// sp weights are read-once streams (268MB total): non-temporal loads keep the
// resident basis tables / h mirrors hot in the 192MB L2.
__global__ void mix_kernel(const float* __restrict__ xc, const float* __restrict__ xs,
                           const float* __restrict__ wc, const float* __restrict__ ws,
                           bf16* fc2, bf16* fs2) {
  int t = blockIdx.x * blockDim.x + threadIdx.x;  // 128*512 threads
  int k = t & (KM - 1);
  int o = t >> 9;
  float fc = 0.f, fs = 0.f;
  for (int i = 0; i < CC; ++i) {
    float a = xc[(size_t)i * KM + k];
    float b = xs[(size_t)i * KM + k];
    float wcv = __builtin_nontemporal_load(&wc[((size_t)i * CC + o) * KM + k]);
    float wsv = __builtin_nontemporal_load(&ws[((size_t)i * CC + o) * KM + k]);
    fc += a * wcv - b * wsv;
    fs += b * wcv + a * wsv;
  }
  fc2[(size_t)o * KM + k] = (bf16)( 2.f * fc);
  fs2[(size_t)o * KM + k] = (bf16)(-2.f * fs);
}

// backward transform + pointwise path + epilogue.
// h' = fc2@bc^T + fs2@bs^T + w_w@h + f0 (x) mask + w_b ; gelu unless last layer.
// wave computes 16x64 (4 tiles): 8 x 256 = 2048 waves. Software-pipelined.
__global__ void gemm_bwd_kernel(const bf16* __restrict__ fc2, const bf16* __restrict__ fs2,
                                const bf16* __restrict__ bc_nm, const bf16* __restrict__ bs_nm,
                                const bf16* __restrict__ wwbf, const bf16* __restrict__ h_nc,
                                const float* __restrict__ f0, const float* __restrict__ wb,
                                const float* __restrict__ mask,
                                float* ho32, bf16* ho_cn, bf16* ho_nc, int do_gelu) {
  int wave = (blockIdx.x * blockDim.x + threadIdx.x) >> 5;
  int lane = threadIdx.x & 31;
  int m0 = (wave >> 8) * 16;
  int n0 = (wave & 255) * 64;
  v8f acc0 = {}, acc1 = {}, acc2 = {}, acc3 = {};

  // ---- spectral part: K = KM, pipelined ----
  v16bf aC = load_a_frag(fc2, KM, m0, 0, lane);
  v16bf aS = load_a_frag(fs2, KM, m0, 0, lane);
  v16bf bC0 = load_bt_frag(bc_nm, KM, 0, n0,      lane);
  v16bf bC1 = load_bt_frag(bc_nm, KM, 0, n0 + 16, lane);
  v16bf bC2 = load_bt_frag(bc_nm, KM, 0, n0 + 32, lane);
  v16bf bC3 = load_bt_frag(bc_nm, KM, 0, n0 + 48, lane);
  v16bf bS0 = load_bt_frag(bs_nm, KM, 0, n0,      lane);
  v16bf bS1 = load_bt_frag(bs_nm, KM, 0, n0 + 16, lane);
  v16bf bS2 = load_bt_frag(bs_nm, KM, 0, n0 + 32, lane);
  v16bf bS3 = load_bt_frag(bs_nm, KM, 0, n0 + 48, lane);
  for (int k0 = 32; k0 < KM; k0 += 32) {
    v16bf aCN = load_a_frag(fc2, KM, m0, k0, lane);
    v16bf aSN = load_a_frag(fs2, KM, m0, k0, lane);
    v16bf nC0 = load_bt_frag(bc_nm, KM, k0, n0,      lane);
    v16bf nC1 = load_bt_frag(bc_nm, KM, k0, n0 + 16, lane);
    v16bf nC2 = load_bt_frag(bc_nm, KM, k0, n0 + 32, lane);
    v16bf nC3 = load_bt_frag(bc_nm, KM, k0, n0 + 48, lane);
    v16bf nS0 = load_bt_frag(bs_nm, KM, k0, n0,      lane);
    v16bf nS1 = load_bt_frag(bs_nm, KM, k0, n0 + 16, lane);
    v16bf nS2 = load_bt_frag(bs_nm, KM, k0, n0 + 32, lane);
    v16bf nS3 = load_bt_frag(bs_nm, KM, k0, n0 + 48, lane);
    acc0 = wmma_bf16(aC, bC0, acc0);
    acc1 = wmma_bf16(aC, bC1, acc1);
    acc2 = wmma_bf16(aC, bC2, acc2);
    acc3 = wmma_bf16(aC, bC3, acc3);
    acc0 = wmma_bf16(aS, bS0, acc0);
    acc1 = wmma_bf16(aS, bS1, acc1);
    acc2 = wmma_bf16(aS, bS2, acc2);
    acc3 = wmma_bf16(aS, bS3, acc3);
    aC = aCN; aS = aSN;
    bC0 = nC0; bC1 = nC1; bC2 = nC2; bC3 = nC3;
    bS0 = nS0; bS1 = nS1; bS2 = nS2; bS3 = nS3;
  }
  // last spectral step overlapped with first pointwise-phase prefetch
  v16bf aW  = load_a_frag(wwbf, CC, m0, 0, lane);
  v16bf bW0 = load_bt_frag(h_nc, CC, 0, n0,      lane);
  v16bf bW1 = load_bt_frag(h_nc, CC, 0, n0 + 16, lane);
  v16bf bW2 = load_bt_frag(h_nc, CC, 0, n0 + 32, lane);
  v16bf bW3 = load_bt_frag(h_nc, CC, 0, n0 + 48, lane);
  acc0 = wmma_bf16(aC, bC0, acc0);
  acc1 = wmma_bf16(aC, bC1, acc1);
  acc2 = wmma_bf16(aC, bC2, acc2);
  acc3 = wmma_bf16(aC, bC3, acc3);
  acc0 = wmma_bf16(aS, bS0, acc0);
  acc1 = wmma_bf16(aS, bS1, acc1);
  acc2 = wmma_bf16(aS, bS2, acc2);
  acc3 = wmma_bf16(aS, bS3, acc3);

  // ---- pointwise part: K = CC, pipelined ----
  for (int k0 = 32; k0 < CC; k0 += 32) {
    v16bf aWN = load_a_frag(wwbf, CC, m0, k0, lane);
    v16bf nW0 = load_bt_frag(h_nc, CC, k0, n0,      lane);
    v16bf nW1 = load_bt_frag(h_nc, CC, k0, n0 + 16, lane);
    v16bf nW2 = load_bt_frag(h_nc, CC, k0, n0 + 32, lane);
    v16bf nW3 = load_bt_frag(h_nc, CC, k0, n0 + 48, lane);
    acc0 = wmma_bf16(aW, bW0, acc0);
    acc1 = wmma_bf16(aW, bW1, acc1);
    acc2 = wmma_bf16(aW, bW2, acc2);
    acc3 = wmma_bf16(aW, bW3, acc3);
    aW = aWN; bW0 = nW0; bW1 = nW1; bW2 = nW2; bW3 = nW3;
  }
  acc0 = wmma_bf16(aW, bW0, acc0);
  acc1 = wmma_bf16(aW, bW1, acc1);
  acc2 = wmma_bf16(aW, bW2, acc2);
  acc3 = wmma_bf16(aW, bW3, acc3);

  // ---- epilogue ----
  int rb = m0 + ((lane & 16) ? 8 : 0);
  v8f accs[4] = {acc0, acc1, acc2, acc3};
#pragma unroll
  for (int t = 0; t < 4; ++t) {
    int col = n0 + 16 * t + (lane & 15);
    float mk = mask[col];
    v8bf packed;
#pragma unroll
    for (int j = 0; j < 8; ++j) {
      int row = rb + j;
      float v = accs[t][j] + f0[row] * mk + wb[row];
      if (do_gelu) v = gelu_f(v);
      ho32[(size_t)row * NN + col]  = v;
      ho_cn[(size_t)row * NN + col] = (bf16)v;
      packed[j] = (bf16)v;
    }
    *(v8bf*)(ho_nc + (size_t)col * CC + rb) = packed;  // 16B store, aligned
  }
}

// fc1: g[f,n] = gelu(fc1_w[f,:] @ h[:,n] + fc1_b[f]) ; wave computes 16x64, pipelined.
__global__ void fc1_kernel(const bf16* __restrict__ fc1bf, const bf16* __restrict__ h_nc,
                           const float* __restrict__ fc1b, float* g32) {
  int wave = (blockIdx.x * blockDim.x + threadIdx.x) >> 5;
  int lane = threadIdx.x & 31;
  int m0 = (wave >> 8) * 16;
  int n0 = (wave & 255) * 64;
  v8f acc0 = {}, acc1 = {}, acc2 = {}, acc3 = {};

  v16bf a  = load_a_frag(fc1bf, CC, m0, 0, lane);
  v16bf b0 = load_bt_frag(h_nc, CC, 0, n0,      lane);
  v16bf b1 = load_bt_frag(h_nc, CC, 0, n0 + 16, lane);
  v16bf b2 = load_bt_frag(h_nc, CC, 0, n0 + 32, lane);
  v16bf b3 = load_bt_frag(h_nc, CC, 0, n0 + 48, lane);
  for (int k0 = 32; k0 < CC; k0 += 32) {
    v16bf aN = load_a_frag(fc1bf, CC, m0, k0, lane);
    v16bf n0f = load_bt_frag(h_nc, CC, k0, n0,      lane);
    v16bf n1f = load_bt_frag(h_nc, CC, k0, n0 + 16, lane);
    v16bf n2f = load_bt_frag(h_nc, CC, k0, n0 + 32, lane);
    v16bf n3f = load_bt_frag(h_nc, CC, k0, n0 + 48, lane);
    acc0 = wmma_bf16(a, b0, acc0);
    acc1 = wmma_bf16(a, b1, acc1);
    acc2 = wmma_bf16(a, b2, acc2);
    acc3 = wmma_bf16(a, b3, acc3);
    a = aN; b0 = n0f; b1 = n1f; b2 = n2f; b3 = n3f;
  }
  acc0 = wmma_bf16(a, b0, acc0);
  acc1 = wmma_bf16(a, b1, acc1);
  acc2 = wmma_bf16(a, b2, acc2);
  acc3 = wmma_bf16(a, b3, acc3);

  int rb = m0 + ((lane & 16) ? 8 : 0);
  v8f accs[4] = {acc0, acc1, acc2, acc3};
#pragma unroll
  for (int t = 0; t < 4; ++t) {
    int col = n0 + 16 * t + (lane & 15);
#pragma unroll
    for (int j = 0; j < 8; ++j) {
      int row = rb + j;
      g32[(size_t)row * NN + col] = gelu_f(accs[t][j] + fc1b[row]);
    }
  }
}

// out[n] = (sum_f g[f,n]*fc2_w[f] + fc2_b) * mask[n]
__global__ void final_kernel(const float* __restrict__ g32, const float* __restrict__ fc2w,
                             const float* __restrict__ fc2b, const float* __restrict__ mask,
                             float* out) {
  int n = blockIdx.x * blockDim.x + threadIdx.x;
  float s = fc2b[0];
#pragma unroll 4
  for (int f = 0; f < CC; ++f) s += g32[(size_t)f * NN + n] * fc2w[f];
  out[n] = s * mask[n];
}

// ---------------- host launcher ----------------

extern "C" void kernel_launch(void* const* d_in, const int* in_sizes, int n_in,
                              void* d_out, int out_size, void* d_ws, size_t ws_size,
                              hipStream_t stream) {
  const float* x         = (const float*)d_in[0];
  const float* node_mask = (const float*)d_in[1];
  const float* nodes     = (const float*)d_in[2];
  const float* wts       = (const float*)d_in[3];
  const float* modes     = (const float*)d_in[4];
  const float* fc0_w     = (const float*)d_in[5];
  const float* fc0_b     = (const float*)d_in[6];
  const float* sp_wc     = (const float*)d_in[7];
  const float* sp_ws     = (const float*)d_in[8];
  const float* sp_w0     = (const float*)d_in[9];
  const float* w_w       = (const float*)d_in[10];
  const float* w_b       = (const float*)d_in[11];
  const float* fc1_w     = (const float*)d_in[12];
  const float* fc1_b     = (const float*)d_in[13];
  const float* fc2_w     = (const float*)d_in[14];
  const float* fc2_b     = (const float*)d_in[15];
  float* out = (float*)d_out;

  char* p = (char*)d_ws;
  auto alloc = [&](size_t bytes) -> char* {
    char* r = p;
    p += (bytes + 255) & ~(size_t)255;
    return r;
  };
  bf16* bc_nm = (bf16*)alloc((size_t)NN * KM * 2);
  bf16* bs_nm = (bf16*)alloc((size_t)NN * KM * 2);
  bf16* wbcT  = (bf16*)alloc((size_t)KM * NN * 2);
  bf16* wbsnT = (bf16*)alloc((size_t)KM * NN * 2);
  float* hA32 = (float*)alloc((size_t)CC * NN * 4);
  float* hB32 = (float*)alloc((size_t)CC * NN * 4);
  bf16*  hAcn = (bf16*)alloc((size_t)CC * NN * 2);
  bf16*  hBcn = (bf16*)alloc((size_t)CC * NN * 2);
  bf16*  hAnc = (bf16*)alloc((size_t)NN * CC * 2);
  bf16*  hBnc = (bf16*)alloc((size_t)NN * CC * 2);
  float* xc32 = (float*)alloc((size_t)CC * KM * 4);
  float* xs32 = (float*)alloc((size_t)CC * KM * 4);
  bf16*  fc2b_ = (bf16*)alloc((size_t)CC * KM * 2);
  bf16*  fs2b_ = (bf16*)alloc((size_t)CC * KM * 2);
  float* x0v   = (float*)alloc(CC * 4);
  float* f0v   = (float*)alloc(CC * 4);
  bf16*  wwbf  = (bf16*)alloc((size_t)CC * CC * 2);
  bf16*  fc1bf = (bf16*)alloc((size_t)CC * CC * 2);
  float* g32   = (float*)alloc((size_t)CC * NN * 4);

  // 1) basis tables (both layouts, weights/signs folded in)
  basis_kernel<<<dim3(KM / 16, NN / 16), dim3(16, 16), 0, stream>>>(
      nodes, modes, node_mask, wts, bc_nm, bs_nm, wbcT, wbsnT);

  // 2) lift
  lift_kernel<<<NN / 256, 256, 0, stream>>>(x, fc0_w, fc0_b, hA32, hAcn, hAnc);

  // one-off weight conversion for fc1
  cvt_bf16_kernel<<<(CC * CC + 255) / 256, 256, 0, stream>>>(fc1_w, fc1bf, CC * CC);

  float* hc32 = hA32; bf16* hccn = hAcn; bf16* hcnc = hAnc;
  float* hn32 = hB32; bf16* hncn = hBcn; bf16* hnnc = hBnc;

  // 3) spectral layers
  for (int l = 0; l < LL; ++l) {
    const float* wcl = sp_wc + (size_t)l * CC * CC * KM;
    const float* wsl = sp_ws + (size_t)l * CC * CC * KM;
    const float* w0l = sp_w0 + (size_t)l * CC * CC;
    const float* wwl = w_w  + (size_t)l * CC * CC;
    const float* wbl = w_b  + (size_t)l * CC;

    x0_kernel<<<CC, 256, 0, stream>>>(hc32, wts, node_mask, x0v);
    f0_kernel<<<1, CC, 0, stream>>>(x0v, w0l, f0v);
    cvt_bf16_kernel<<<(CC * CC + 255) / 256, 256, 0, stream>>>(wwl, wwbf, CC * CC);

    gemm_fwd_kernel<<<16, 256, 0, stream>>>(hccn, wbcT, wbsnT, xc32, xs32);
    mix_kernel<<<(CC * KM) / 256, 256, 0, stream>>>(xc32, xs32, wcl, wsl, fc2b_, fs2b_);
    gemm_bwd_kernel<<<256, 256, 0, stream>>>(fc2b_, fs2b_, bc_nm, bs_nm, wwbf, hcnc,
                                             f0v, wbl, node_mask, hn32, hncn, hnnc,
                                             (l != LL - 1) ? 1 : 0);
    // swap
    float* t32 = hc32; hc32 = hn32; hn32 = t32;
    bf16*  tcn = hccn; hccn = hncn; hncn = tcn;
    bf16*  tnc = hcnc; hcnc = hnnc; hnnc = tnc;
  }

  // 4) head
  fc1_kernel<<<256, 256, 0, stream>>>(fc1bf, hcnc, fc1_b, g32);
  final_kernel<<<NN / 256, 256, 0, stream>>>(g32, fc2_w, fc2_b, node_mask, out);
  (void)in_sizes; (void)n_in; (void)out_size; (void)ws_size;
}
